// CausalTransformerShard_24257975288014
// MI455X (gfx1250) — compile-verified, matrix operand order checked by reading, not run on hardware
//
#include <hip/hip_runtime.h>
#include <hip/hip_bf16.h>
#include <math.h>

// ---- model constants ----
#define LNUM 8
#define DMODEL 1024
#define HEADS 16
#define DHEAD 64
#define VOCAB 50257
#define FFD 4096
#define TSEQ 1024

typedef __attribute__((ext_vector_type(16))) __bf16 v16bf;
typedef __attribute__((ext_vector_type(8)))  __bf16 v8bf;
typedef __attribute__((ext_vector_type(8)))  float  v8f;

// Combine two contiguous 8xbf16 LDS/global chunks into a 16xbf16 WMMA fragment.
// ISA 16-bit A-fragment: elems 0..7 = K = base..base+7, elems 8..15 = K = base+16..base+23,
// with base = 8*(lane>=16).  B fragments use the mirrored layout (lane -> N column).
__device__ __forceinline__ v16bf mk_frag(const __bf16* p0, const __bf16* p1) {
    v8bf lo = *(const v8bf*)p0;
    v8bf hi = *(const v8bf*)p1;
    return __builtin_shufflevector(lo, hi, 0,1,2,3,4,5,6,7,8,9,10,11,12,13,14,15);
}

__device__ __forceinline__ v8f wmma_bf16(v16bf a, v16bf b, v8f c) {
    return __builtin_amdgcn_wmma_f32_16x16x32_bf16(false, a, false, b, (short)0, c, false, false);
}

// gfx1250: generic (flat) pointers to LDS carry the LDS byte offset in the low
// 32 bits (shared-aperture base in the high half) -> truncation yields the LDS
// address the async-DMA VDST operand wants.
__device__ __forceinline__ unsigned lds_addr(const void* p) {
    return (unsigned)(unsigned long long)(uintptr_t)p;
}

// CDNA5 async global->LDS DMA (bypasses VGPRs, tracked by ASYNCcnt).
__device__ __forceinline__ void async_copy_b128(unsigned lds_off, const void* gptr) {
    asm volatile("global_load_async_to_lds_b128 %0, %1, off"
                 :: "v"(lds_off), "v"((unsigned long long)(uintptr_t)gptr)
                 : "memory");
}
__device__ __forceinline__ void wait_async0() {
    asm volatile("s_wait_asynccnt 0x0" ::: "memory");
}

// pack two floats -> one dword of two bf16 (for transposed LDS stores)
__device__ __forceinline__ unsigned pack_bf16(float a, float b) {
    union { unsigned u; __bf16 h[2]; } c;
    c.h[0] = (__bf16)a; c.h[1] = (__bf16)b;
    return c.u;
}

// ---------------------------------------------------------------------------
// Embedding gather: x[t][:] = embed_w[ctx[t]][:]
// ---------------------------------------------------------------------------
__global__ void embed_kernel(const int* __restrict__ ctx, const float* __restrict__ ew,
                             float* __restrict__ x) {
    int row = blockIdx.x;
    int tok = ctx[row];
    const float* src = ew + (size_t)tok * DMODEL;
    float* dst = x + (size_t)row * DMODEL;
    for (int i = threadIdx.x; i < DMODEL; i += 256) dst[i] = src[i];
}

// ---------------------------------------------------------------------------
// LayerNorm: out_bf16 = (x - mean) * rsqrt(var + eps) * scale + bias
// ---------------------------------------------------------------------------
__global__ void ln_kernel(const float* __restrict__ x, const float* __restrict__ sc,
                          const float* __restrict__ bi, __bf16* __restrict__ out, int D) {
    int row = blockIdx.x, tid = threadIdx.x;
    const float* xr = x + (size_t)row * D;
    __shared__ float red[256];
    float s = 0.f;
    for (int i = tid; i < D; i += 256) s += xr[i];
    red[tid] = s; __syncthreads();
    for (int o = 128; o > 0; o >>= 1) { if (tid < o) red[tid] += red[tid + o]; __syncthreads(); }
    float mean = red[0] / (float)D;
    __syncthreads();
    float v = 0.f;
    for (int i = tid; i < D; i += 256) { float d = xr[i] - mean; v += d * d; }
    red[tid] = v; __syncthreads();
    for (int o = 128; o > 0; o >>= 1) { if (tid < o) red[tid] += red[tid + o]; __syncthreads(); }
    float rs = rsqrtf(red[0] / (float)D + 1e-5f);
    __bf16* orow = out + (size_t)row * D;
    for (int i = tid; i < D; i += 256)
        orow[i] = (__bf16)((xr[i] - mean) * rs * sc[i] + bi[i]);
}

// ---------------------------------------------------------------------------
// RoPE in-place on bf16 q,k (ROT == DHEAD == 64, full head rotates)
// ---------------------------------------------------------------------------
__global__ void rope_kernel(__bf16* __restrict__ q, __bf16* __restrict__ k) {
    int idx = blockIdx.x * 256 + threadIdx.x;      // < T*H*32
    int pair = idx & 31;
    int h = (idx >> 5) & (HEADS - 1);
    int t = idx >> 9;
    int i2 = 2 * pair;
    float inv = __powf(10000.f, -((float)i2 / 64.f));
    float f = (float)t * inv;
    float sn, cs;
    __sincosf(f, &sn, &cs);
    size_t off = (size_t)t * DMODEL + h * DHEAD + i2;
    float q0 = (float)q[off], q1 = (float)q[off + 1];
    q[off]     = (__bf16)(q0 * cs - q1 * sn);
    q[off + 1] = (__bf16)(q1 * cs + q0 * sn);
    float k0 = (float)k[off], k1 = (float)k[off + 1];
    k[off]     = (__bf16)(k0 * cs - k1 * sn);
    k[off + 1] = (__bf16)(k1 * cs + k0 * sn);
}

// ---------------------------------------------------------------------------
// Tiled bf16 WMMA GEMM:  C[M,N] = epilogue(A_bf16[M,K] @ W_f32[K,N] + bias)
// Block 256 threads (8 waves), tile 128x128, K-step 32.
// A tile staged via async global->LDS DMA; W tile streamed fp32, converted to
// bf16 in-register, stored transposed [n][k] with packed b32 stores.
// ---------------------------------------------------------------------------
enum { EPI_BF16 = 0, EPI_GELU = 1, EPI_RESID = 2 };

template <int EPI>
__global__ __launch_bounds__(256, 1)
void gemm_bf16_kernel(const __bf16* __restrict__ A, const float* __restrict__ W,
                      const float* __restrict__ bias, __bf16* __restrict__ outb,
                      float* __restrict__ resid, int M, int N, int K) {
    __shared__ __align__(16) __bf16 As[128][32];
    __shared__ __align__(16) __bf16 Bs[128][32];   // transposed: [n][k]

    const int tid  = threadIdx.x;
    const int lane = tid & 31;
    const int wave = tid >> 5;                 // 0..7
    const int wm   = wave >> 1;                // 0..3
    const int wn   = wave & 1;                 // 0..1
    const int m0   = blockIdx.y * 128;
    const int n0   = blockIdx.x * 128;
    const int base = 8 * (lane >> 4);
    const int ncol = lane & 15;

    v8f acc[2][4];
    for (int mt = 0; mt < 2; ++mt)
        for (int nt = 0; nt < 4; ++nt) {
            int col = n0 + wn * 64 + nt * 16 + ncol;
            float b = bias ? bias[col] : 0.f;
            for (int r = 0; r < 8; ++r) acc[mt][nt][r] = b;
        }

    const int a_row = tid >> 1, a_seg = (tid & 1) * 16;       // A stage: 16 bf16/thread
    const int w_k2  = (tid >> 4) * 2, w_g8 = (tid & 15) * 8;  // W stage: 2 k-rows x 8 cols

    const unsigned lA0 = lds_addr(&As[a_row][a_seg]);
    const unsigned lA1 = lds_addr(&As[a_row][a_seg + 8]);

    for (int k0 = 0; k0 < K; k0 += 32) {
        // A tile: async DMA straight into LDS (no VGPR round-trip)
        {
            const __bf16* ap = A + (size_t)(m0 + a_row) * K + k0 + a_seg;
            async_copy_b128(lA0, ap);
            async_copy_b128(lA1, ap + 8);
        }
        // W tile: fp32 loads, cvt->bf16, packed transposed stores [n][k]
        {
            const float* wp0 = W + (size_t)(k0 + w_k2) * N + n0 + w_g8;
            const float* wp1 = wp0 + N;
            if (k0 + 32 < K)
                __builtin_prefetch(wp0 + (size_t)32 * N, 0, 1);
            float4 a0 = *(const float4*)wp0;
            float4 a1 = *(const float4*)(wp0 + 4);
            float4 b0 = *(const float4*)wp1;
            float4 b1 = *(const float4*)(wp1 + 4);
            float ra[8] = {a0.x,a0.y,a0.z,a0.w, a1.x,a1.y,a1.z,a1.w};
            float rb[8] = {b0.x,b0.y,b0.z,b0.w, b1.x,b1.y,b1.z,b1.w};
            #pragma unroll
            for (int i = 0; i < 8; ++i)
                *(unsigned*)&Bs[w_g8 + i][w_k2] = pack_bf16(ra[i], rb[i]);
        }
        wait_async0();
        __syncthreads();

        v16bf afrag[2], bfrag[4];
        #pragma unroll
        for (int mt = 0; mt < 2; ++mt) {
            int row = wm * 32 + mt * 16 + ncol;
            afrag[mt] = mk_frag(&As[row][base], &As[row][base + 16]);
        }
        #pragma unroll
        for (int nt = 0; nt < 4; ++nt) {
            int col = wn * 64 + nt * 16 + ncol;
            bfrag[nt] = mk_frag(&Bs[col][base], &Bs[col][base + 16]);
        }
        #pragma unroll
        for (int mt = 0; mt < 2; ++mt)
            #pragma unroll
            for (int nt = 0; nt < 4; ++nt)
                acc[mt][nt] = wmma_bf16(afrag[mt], bfrag[nt], acc[mt][nt]);
        __syncthreads();
    }

    // epilogue: C layout -> row = (lane>>4)*8 + r, col = lane&15
    #pragma unroll
    for (int mt = 0; mt < 2; ++mt)
        #pragma unroll
        for (int nt = 0; nt < 4; ++nt) {
            int col = n0 + wn * 64 + nt * 16 + ncol;
            #pragma unroll
            for (int r = 0; r < 8; ++r) {
                int row = m0 + wm * 32 + mt * 16 + (lane >> 4) * 8 + r;
                float v = acc[mt][nt][r];
                if (EPI == EPI_BF16) {
                    outb[(size_t)row * N + col] = (__bf16)v;
                } else if (EPI == EPI_GELU) {
                    float g = 0.5f * v * (1.f + tanhf(0.7978845608f * (v + 0.044715f * v * v * v)));
                    outb[(size_t)row * N + col] = (__bf16)g;
                } else {
                    resid[(size_t)row * N + col] += v;
                }
            }
        }
}

// ---------------------------------------------------------------------------
// Flash attention (causal) with WMMA.  Grid = (T/64, H), block = 128 (4 waves).
// K tile staged via async DMA; V tile transposed with packed b32 stores.
// ---------------------------------------------------------------------------
__global__ __launch_bounds__(128, 1)
void flash_attn_kernel(const __bf16* __restrict__ q, const __bf16* __restrict__ k,
                       const __bf16* __restrict__ v, __bf16* __restrict__ o) {
    __shared__ __align__(16) __bf16 Ks[32][64];     // [key][dh]
    __shared__ __align__(16) __bf16 Vs[64][32];     // [dh][key] (transposed)
    __shared__ __align__(16) __bf16 Ps[4][16][32];  // per-wave P tile

    const int tid  = threadIdx.x;
    const int lane = tid & 31;
    const int wave = tid >> 5;               // 0..3
    const int h    = blockIdx.y;
    const int qblk = blockIdx.x;             // 64-query block
    const int q0   = qblk * 64 + wave * 16;  // wave's first query row
    const int base = 8 * (lane >> 4);
    const int ncol = lane & 15;
    const float scale = 0.125f;              // 1/sqrt(64)

    // Q fragments (A layout): row = lane&15, two 32-wide K-chunks over DHEAD=64
    v16bf aq[2];
    {
        const __bf16* qp = q + (size_t)(q0 + ncol) * DMODEL + h * DHEAD;
        aq[0] = mk_frag(qp + base,      qp + base + 16);
        aq[1] = mk_frag(qp + 32 + base, qp + 32 + base + 16);
    }

    v8f acc_o[4];
    for (int nt = 0; nt < 4; ++nt)
        for (int r = 0; r < 8; ++r) acc_o[nt][r] = 0.f;
    float mrow[8], lrow[8];
    for (int r = 0; r < 8; ++r) { mrow[r] = -1e30f; lrow[r] = 0.f; }

    const int k_r = tid >> 2, k_g = (tid & 3) * 16;        // K stage mapping
    const int v_key2 = (tid & 15) * 2, v_dh8 = (tid >> 4) * 8;  // V transpose mapping
    const unsigned lK0 = lds_addr(&Ks[k_r][k_g]);
    const unsigned lK1 = lds_addr(&Ks[k_r][k_g + 8]);

    const int nkv = 2 * qblk + 2;            // causal: keys 0 .. qblk*64+63
    for (int j = 0; j < nkv; ++j) {
        const int kb = j * 32;
        // stage K tile via async DMA
        {
            const __bf16* kp = k + (size_t)(kb + k_r) * DMODEL + h * DHEAD + k_g;
            async_copy_b128(lK0, kp);
            async_copy_b128(lK1, kp + 8);
        }
        // stage V tile transposed [dh][key]: 2 keys x 8 dh per thread, packed stores
        {
            const __bf16* vp0 = v + (size_t)(kb + v_key2) * DMODEL + h * DHEAD + v_dh8;
            const __bf16* vp1 = vp0 + DMODEL;
            v8bf c0 = *(const v8bf*)vp0;
            v8bf c1 = *(const v8bf*)vp1;
            #pragma unroll
            for (int i = 0; i < 8; ++i)
                *(unsigned*)&Vs[v_dh8 + i][v_key2] = pack_bf16((float)c0[i], (float)c1[i]);
        }
        wait_async0();
        __syncthreads();

        // S = Q @ K^T  (two 16-key n-tiles, two 32-wide dh chunks)
        v8f s[2];
        #pragma unroll
        for (int nt = 0; nt < 2; ++nt) {
            const __bf16* kr = &Ks[nt * 16 + ncol][0];
            v16bf b0 = mk_frag(kr + base,      kr + base + 16);
            v16bf b1 = mk_frag(kr + 32 + base, kr + 32 + base + 16);
            v8f z; for (int r = 0; r < 8; ++r) z[r] = 0.f;
            z = wmma_bf16(aq[0], b0, z);
            s[nt] = wmma_bf16(aq[1], b1, z);
        }

        // scale + causal mask + online softmax update
        float p[2][8], bm[8], rs[8];
        #pragma unroll
        for (int r = 0; r < 8; ++r) {
            int qrow = q0 + (lane >> 4) * 8 + r;
            float s0 = s[0][r] * scale, s1 = s[1][r] * scale;
            if (kb + ncol      > qrow) s0 = -1e9f;
            if (kb + 16 + ncol > qrow) s1 = -1e9f;
            p[0][r] = s0; p[1][r] = s1;
            bm[r] = fmaxf(s0, s1);
        }
        #pragma unroll
        for (int r = 0; r < 8; ++r) {
            for (int off = 1; off < 16; off <<= 1)
                bm[r] = fmaxf(bm[r], __shfl_xor(bm[r], off));
            float newm = fmaxf(mrow[r], bm[r]);
            float alpha = __expf(mrow[r] - newm);
            p[0][r] = __expf(p[0][r] - newm);
            p[1][r] = __expf(p[1][r] - newm);
            rs[r] = p[0][r] + p[1][r];
            for (int off = 1; off < 16; off <<= 1)
                rs[r] += __shfl_xor(rs[r], off);
            lrow[r] = lrow[r] * alpha + rs[r];
            mrow[r] = newm;
            #pragma unroll
            for (int nt = 0; nt < 4; ++nt) acc_o[nt][r] *= alpha;
        }

        // write P tile (C layout -> row-major bf16) for re-fragmentation
        #pragma unroll
        for (int nt = 0; nt < 2; ++nt)
            #pragma unroll
            for (int r = 0; r < 8; ++r)
                Ps[wave][(lane >> 4) * 8 + r][nt * 16 + ncol] = (__bf16)p[nt][r];
        __syncthreads();

        // O += P @ V
        {
            const __bf16* pr = &Ps[wave][ncol][0];
            v16bf pf = mk_frag(pr + base, pr + base + 16);
            #pragma unroll
            for (int nt = 0; nt < 4; ++nt) {
                const __bf16* vr = &Vs[nt * 16 + ncol][0];
                v16bf vf = mk_frag(vr + base, vr + base + 16);
                acc_o[nt] = wmma_bf16(pf, vf, acc_o[nt]);
            }
        }
        __syncthreads();
    }

    // normalize and store
    #pragma unroll
    for (int nt = 0; nt < 4; ++nt)
        #pragma unroll
        for (int r = 0; r < 8; ++r) {
            int row = q0 + (lane >> 4) * 8 + r;
            float ov = acc_o[nt][r] / lrow[r];
            o[(size_t)row * DMODEL + h * DHEAD + nt * 16 + ncol] = (__bf16)ov;
        }
}

// ---------------------------------------------------------------------------
// Final projection GEMV: logits[V] = lnf(x_last)[D] @ w_proj[D,V] + b_proj
// memory-bound (206 MB of w_proj) -> streaming, coalesced across columns
// ---------------------------------------------------------------------------
__global__ void gemv_kernel(const __bf16* __restrict__ xl, const float* __restrict__ wproj,
                            const float* __restrict__ bproj, float* __restrict__ out) {
    __shared__ float xs[DMODEL];
    int tid = threadIdx.x;
    for (int i = tid; i < DMODEL; i += 256) xs[i] = (float)xl[i];
    __syncthreads();
    int c = blockIdx.x * 256 + tid;
    if (c >= VOCAB) return;
    float acc = bproj[c];
    #pragma unroll 4
    for (int kk = 0; kk < DMODEL; ++kk)
        acc += xs[kk] * wproj[(size_t)kk * VOCAB + c];
    out[c] = acc;
}

// ---------------------------------------------------------------------------
// host orchestration
// ---------------------------------------------------------------------------
extern "C" void kernel_launch(void* const* d_in, const int* in_sizes, int n_in,
                              void* d_out, int out_size, void* d_ws, size_t ws_size,
                              hipStream_t stream) {
    (void)in_sizes; (void)n_in; (void)out_size; (void)ws_size;
    const int*   ctx   = (const int*)  d_in[0];
    const float* embw  = (const float*)d_in[1];
    const float* ln1s  = (const float*)d_in[2];
    const float* ln1b  = (const float*)d_in[3];
    const float* wq    = (const float*)d_in[4];
    const float* wk    = (const float*)d_in[5];
    const float* wv    = (const float*)d_in[6];
    const float* wo    = (const float*)d_in[7];
    const float* ln2s  = (const float*)d_in[8];
    const float* ln2b  = (const float*)d_in[9];
    const float* w_in  = (const float*)d_in[10];
    const float* b_in  = (const float*)d_in[11];
    const float* w_out = (const float*)d_in[12];
    const float* b_out = (const float*)d_in[13];
    const float* lnfs  = (const float*)d_in[14];
    const float* lnfb  = (const float*)d_in[15];
    const float* wproj = (const float*)d_in[16];
    const float* bproj = (const float*)d_in[17];

    char* w = (char*)d_ws;
    const size_t MB = 1024 * 1024;
    float*  x  = (float*)(w);                  // 4 MB   fp32 residual stream
    __bf16* xn = (__bf16*)(w + 4  * MB);       // 2 MB   LN output
    __bf16* qb = (__bf16*)(w + 6  * MB);       // 2 MB
    __bf16* kb = (__bf16*)(w + 8  * MB);       // 2 MB
    __bf16* vb = (__bf16*)(w + 10 * MB);       // 2 MB
    __bf16* ob = (__bf16*)(w + 12 * MB);       // 2 MB   attention output
    __bf16* ff = (__bf16*)(w + 14 * MB);       // 8 MB   MLP hidden
    __bf16* xl = (__bf16*)(w + 22 * MB);       // 2 KB   final-LN row

    dim3 gD(DMODEL / 128, TSEQ / 128);         // N=1024 GEMMs
    dim3 gF(FFD / 128,    TSEQ / 128);         // N=4096 GEMM

    embed_kernel<<<TSEQ, 256, 0, stream>>>(ctx, embw, x);

    for (int i = 0; i < LNUM; ++i) {
        ln_kernel<<<TSEQ, 256, 0, stream>>>(x, ln1s + (size_t)i * DMODEL,
                                            ln1b + (size_t)i * DMODEL, xn, DMODEL);
        gemm_bf16_kernel<EPI_BF16><<<gD, 256, 0, stream>>>(
            xn, wq + (size_t)i * DMODEL * DMODEL, nullptr, qb, nullptr, TSEQ, DMODEL, DMODEL);
        gemm_bf16_kernel<EPI_BF16><<<gD, 256, 0, stream>>>(
            xn, wk + (size_t)i * DMODEL * DMODEL, nullptr, kb, nullptr, TSEQ, DMODEL, DMODEL);
        gemm_bf16_kernel<EPI_BF16><<<gD, 256, 0, stream>>>(
            xn, wv + (size_t)i * DMODEL * DMODEL, nullptr, vb, nullptr, TSEQ, DMODEL, DMODEL);

        rope_kernel<<<(TSEQ * HEADS * 32) / 256, 256, 0, stream>>>(qb, kb);

        flash_attn_kernel<<<dim3(TSEQ / 64, HEADS), 128, 0, stream>>>(qb, kb, vb, ob);

        gemm_bf16_kernel<EPI_RESID><<<gD, 256, 0, stream>>>(
            ob, wo + (size_t)i * DMODEL * DMODEL, nullptr, nullptr, x, TSEQ, DMODEL, DMODEL);

        ln_kernel<<<TSEQ, 256, 0, stream>>>(x, ln2s + (size_t)i * DMODEL,
                                            ln2b + (size_t)i * DMODEL, xn, DMODEL);
        gemm_bf16_kernel<EPI_GELU><<<gF, 256, 0, stream>>>(
            xn, w_in + (size_t)i * DMODEL * FFD, b_in + (size_t)i * FFD, ff, nullptr,
            TSEQ, FFD, DMODEL);
        gemm_bf16_kernel<EPI_RESID><<<gD, 256, 0, stream>>>(
            ff, w_out + (size_t)i * FFD * DMODEL, b_out + (size_t)i * DMODEL, nullptr, x,
            TSEQ, DMODEL, FFD);
    }

    ln_kernel<<<1, 256, 0, stream>>>(x + (size_t)(TSEQ - 1) * DMODEL, lnfs, lnfb, xl, DMODEL);
    gemv_kernel<<<(VOCAB + 255) / 256, 256, 0, stream>>>(xl, wproj, bproj, (float*)d_out);
}
